// AttentionModel_77661598646473
// MI455X (gfx1250) — matrix operand order repaired and checked by reading, not compiled, not run on hardware
//
#include <hip/hip_runtime.h>
#include <hip/hip_bf16.h>

typedef __bf16 bf16;
typedef __attribute__((ext_vector_type(16))) __bf16 v16bf;
typedef __attribute__((ext_vector_type(8)))  __bf16 v8bf;
typedef __attribute__((ext_vector_type(8)))  float  v8f;

#define E_DIM 32
#define F_DIM 64
#define S_LEN 256
#define O_DIM 8

struct Bf16x16 { v8bf h[2]; };

__device__ __forceinline__ v16bf frag_from(v8bf a, v8bf b) {
  Bf16x16 t; t.h[0] = a; t.h[1] = b;
  return __builtin_bit_cast(v16bf, t);
}

// A-fragment (16x32 bf16, MxK) from row-major [rows][ldk] source.
// Lane L<16: row L, K {0..7,16..23}; lane L>=16: row L-16, K {8..15,24..31}.
__device__ __forceinline__ v16bf load_A(const bf16* base, int ldk, int row0, int k0) {
  int lane = threadIdx.x & 31;
  int m  = lane & 15;
  int hk = (lane & 16) ? 8 : 0;
  const bf16* p = base + (row0 + m) * ldk + k0 + hk;
  return frag_from(*(const v8bf*)p, *(const v8bf*)(p + 16));
}

// B-fragment (32x16 bf16, KxN) from N-major storage Bt[N][K] (row n of Bt = column n of B).
// Lane L<16: col L, K {0..7,16..23}; lane L>=16: col L-16, K {8..15,24..31}.
__device__ __forceinline__ v16bf load_B(const bf16* bt, int ldk, int n0, int k0) {
  int lane = threadIdx.x & 31;
  int n  = lane & 15;
  int ko = (lane & 16) ? 8 : 0;
  const bf16* p = bt + (n0 + n) * ldk + k0 + ko;
  return frag_from(*(const v8bf*)p, *(const v8bf*)(p + 16));
}

__device__ __forceinline__ v8f wmma_bf16(v16bf a, v16bf b, v8f c) {
  return __builtin_amdgcn_wmma_f32_16x16x32_bf16(false, a, false, b, (short)0, c, false, false);
}

__device__ __forceinline__ v8f splat8(float v) {
  v8f r;
#pragma unroll
  for (int i = 0; i < 8; ++i) r[i] = v;
  return r;
}

__device__ __forceinline__ v8f bias_frag(const float* bias, int n0) {
  return splat8(bias[n0 + (threadIdx.x & 15)]);
}

// D-frag (16x16 f32): lane = column n, VGPR v = row mb+v with mb=(lane<16?0:8).
__device__ __forceinline__ void store_D_bf16(bf16* dst, int ld, int row0, int n0, v8f d) {
  int lane = threadIdx.x & 31;
  int n = lane & 15;
  int mb = (lane & 16) ? 8 : 0;
#pragma unroll
  for (int v = 0; v < 8; ++v) dst[(row0 + mb + v) * ld + n0 + n] = (bf16)d[v];
}

__device__ __forceinline__ void store_D_bf16_T(bf16* dstT, int ldT, int row0, int n0, v8f d) {
  int lane = threadIdx.x & 31;
  int n = lane & 15;
  int mb = (lane & 16) ? 8 : 0;
#pragma unroll
  for (int v = 0; v < 8; ++v) dstT[(n0 + n) * ldT + row0 + mb + v] = (bf16)d[v];
}

__device__ __forceinline__ void add_D_f32(float* dst, int ld, int row0, int n0, v8f d) {
  int lane = threadIdx.x & 31;
  int n = lane & 15;
  int mb = (lane & 16) ? 8 : 0;
#pragma unroll
  for (int v = 0; v < 8; ++v) dst[(row0 + mb + v) * ld + n0 + n] += d[v];
}

// ---- prep kernel: bf16 + transposed copies of the matmul weights into d_ws ----
// layout (bf16 elems): [0,1024) wqT[32][32], [1024,2048) wkT, [2048,3072) wvT,
//                      [3072,5120) w1T[64][32], [5120,7168) w2T[32][64]
__global__ void prep_weights(const float* __restrict__ wq, const float* __restrict__ wk,
                             const float* __restrict__ wv, const float* __restrict__ w1,
                             const float* __restrict__ w2, bf16* __restrict__ ws) {
  int t = threadIdx.x;
  for (int i = t; i < 1024; i += 256) {
    int r = i >> 5, c = i & 31;           // w[r][c], E_in x E_out
    ws[0    + c * 32 + r] = (bf16)wq[i];
    ws[1024 + c * 32 + r] = (bf16)wk[i];
    ws[2048 + c * 32 + r] = (bf16)wv[i];
  }
  for (int i = t; i < 2048; i += 256) {
    int e = i >> 6, f = i & 63;           // w1[e][f], 32x64
    ws[3072 + f * 32 + e] = (bf16)w1[i];
    int f2 = i >> 5, e2 = i & 31;         // w2[f2][e2], 64x32
    ws[5120 + e2 * 64 + f2] = (bf16)w2[i];
  }
}

__device__ __forceinline__ void ln_row(float* tokf, bf16* actb, const float* g,
                                       const float* be, int s) {
  float mu = 0.f;
#pragma unroll
  for (int e = 0; e < E_DIM; ++e) mu += tokf[s * E_DIM + e];
  mu *= (1.f / E_DIM);
  float var = 0.f;
#pragma unroll
  for (int e = 0; e < E_DIM; ++e) {
    float d = tokf[s * E_DIM + e] - mu;
    var += d * d;
  }
  var *= (1.f / E_DIM);
  float r = rsqrtf(var + 1e-5f);
#pragma unroll
  for (int e = 0; e < E_DIM; ++e) {
    float y = (tokf[s * E_DIM + e] - mu) * r * g[e] + be[e];
    tokf[s * E_DIM + e] = y;
    if (actb) actb[s * E_DIM + e] = (bf16)y;
  }
}

__global__ __launch_bounds__(256) void attn_block_kernel(
    const float* __restrict__ x, const float* __restrict__ w_tok,
    const float* __restrict__ b_tok, const float* __restrict__ pos,
    const float* __restrict__ bq, const float* __restrict__ bk,
    const float* __restrict__ bv, const float* __restrict__ b1,
    const float* __restrict__ b2, const float* __restrict__ g1,
    const float* __restrict__ be1, const float* __restrict__ g2,
    const float* __restrict__ be2, const float* __restrict__ wo,
    const float* __restrict__ bo, const bf16* __restrict__ ws,
    float* __restrict__ out) {
  __shared__ alignas(16) float tokf[S_LEN * E_DIM];            // 32 KB residual stream (f32)
  __shared__ alignas(16) bf16  actb[S_LEN * E_DIM];            // 16 KB current activations (bf16, A operand)
  __shared__ alignas(16) bf16  kvbuf[S_LEN * E_DIM * 2];       // 32 KB: K[S][E] + Vt[E][S]  (reused as H[S][F])
  __shared__ alignas(16) bf16  qscr[8 * 16 * E_DIM];           //  8 KB per-wave Q tile scratch
  __shared__ alignas(16) bf16  pscr[8 * 16 * S_LEN];           // 64 KB per-wave softmax-prob scratch
  __shared__ alignas(16) float poolscr[256];
  __shared__ alignas(16) float pooled[E_DIM];

  bf16* kmat = kvbuf;                    // [S][E]
  bf16* vmt  = kvbuf + S_LEN * E_DIM;    // [E][S]
  bf16* hmat = kvbuf;                    // [S][F] (reuse after attention)

  const bf16* WQT = ws;
  const bf16* WKT = ws + 1024;
  const bf16* WVT = ws + 2048;
  const bf16* W1T = ws + 3072;
  const bf16* W2T = ws + 5120;

  const int b    = blockIdx.x;
  const int tid  = threadIdx.x;
  const int wave = tid >> 5;
  bf16* qw = qscr + wave * 16 * E_DIM;
  bf16* pw = pscr + wave * 16 * S_LEN;

  // ---- Stage A: tokens = x*w_tok + b_tok + pos ----
  {
    float xv = x[b * S_LEN + tid];
#pragma unroll 8
    for (int e = 0; e < E_DIM; ++e) {
      float t = xv * w_tok[e] + b_tok[e] + pos[tid * E_DIM + e];
      tokf[tid * E_DIM + e] = t;
      actb[tid * E_DIM + e] = (bf16)t;
    }
  }
  __syncthreads();

  // ---- Stage B: K = tok@wk+bk, Vt = (tok@wv+bv)^T ----
#pragma unroll
  for (int rr = 0; rr < 2; ++rr) {
    int rt = wave * 2 + rr;
    v16bf aT = load_A(actb, E_DIM, rt * 16, 0);
#pragma unroll
    for (int c = 0; c < 2; ++c) {
      v8f kf = wmma_bf16(aT, load_B(WKT, E_DIM, c * 16, 0), bias_frag(bk, c * 16));
      store_D_bf16(kmat, E_DIM, rt * 16, c * 16, kf);
      v8f vf = wmma_bf16(aT, load_B(WVT, E_DIM, c * 16, 0), bias_frag(bv, c * 16));
      store_D_bf16_T(vmt, S_LEN, rt * 16, c * 16, vf);
    }
  }
  __syncthreads();

  // ---- Stage C: attention per row-tile ----
  const float scale = 0.17677669529663687f;  // 1/sqrt(32)
#pragma unroll 1
  for (int rr = 0; rr < 2; ++rr) {
    int rt = wave * 2 + rr;
    v16bf aT = load_A(actb, E_DIM, rt * 16, 0);
#pragma unroll
    for (int c = 0; c < 2; ++c) {
      v8f qf = wmma_bf16(aT, load_B(WQT, E_DIM, c * 16, 0), bias_frag(bq, c * 16));
      store_D_bf16(qw, E_DIM, 0, c * 16, qf);
    }
    asm volatile("s_wait_dscnt 0" ::: "memory");  // wave-local LDS turnaround
    v16bf aQ = load_A(qw, E_DIM, 0, 0);

    v8f sfr[16];
#pragma unroll
    for (int ct = 0; ct < 16; ++ct)
      sfr[ct] = wmma_bf16(aQ, load_B(kmat, E_DIM, ct * 16, 0), splat8(0.f));

    // softmax over the 256 columns (rows live in fixed VGPR slot across 16 lanes)
#pragma unroll
    for (int vv = 0; vv < 8; ++vv) {
      float m = -1e30f;
#pragma unroll
      for (int ct = 0; ct < 16; ++ct) m = fmaxf(m, sfr[ct][vv]);
#pragma unroll
      for (int msk = 1; msk < 16; msk <<= 1) m = fmaxf(m, __shfl_xor(m, msk, 32));
      float sum = 0.f;
#pragma unroll
      for (int ct = 0; ct < 16; ++ct) {
        float p = __expf((sfr[ct][vv] - m) * scale);
        sfr[ct][vv] = p;
        sum += p;
      }
#pragma unroll
      for (int msk = 1; msk < 16; msk <<= 1) sum += __shfl_xor(sum, msk, 32);
      float inv = 1.f / sum;
#pragma unroll
      for (int ct = 0; ct < 16; ++ct) sfr[ct][vv] *= inv;
    }

#pragma unroll
    for (int ct = 0; ct < 16; ++ct) store_D_bf16(pw, S_LEN, 0, ct * 16, sfr[ct]);
    asm volatile("s_wait_dscnt 0" ::: "memory");

    v8f acc0 = splat8(0.f), acc1 = splat8(0.f);
#pragma unroll
    for (int kc = 0; kc < 8; ++kc) {
      v16bf aP = load_A(pw, S_LEN, 0, kc * 32);
      acc0 = wmma_bf16(aP, load_B(vmt, S_LEN, 0, kc * 32), acc0);
      acc1 = wmma_bf16(aP, load_B(vmt, S_LEN, 16, kc * 32), acc1);
    }
    add_D_f32(tokf, E_DIM, rt * 16, 0, acc0);
    add_D_f32(tokf, E_DIM, rt * 16, 16, acc1);
  }
  __syncthreads();

  // ---- LN1 (one row per thread) ----
  ln_row(tokf, actb, g1, be1, tid);
  __syncthreads();

  // ---- Stage D: FFN ----
#pragma unroll
  for (int rr = 0; rr < 2; ++rr) {
    int rt = wave * 2 + rr;
    v16bf aT = load_A(actb, E_DIM, rt * 16, 0);
#pragma unroll
    for (int c = 0; c < 4; ++c) {
      v8f hf = wmma_bf16(aT, load_B(W1T, E_DIM, c * 16, 0), bias_frag(b1, c * 16));
#pragma unroll
      for (int i = 0; i < 8; ++i) {
        float xg = hf[i];
        hf[i] = 0.5f * xg * (1.f + erff(xg * 0.70710678118654752f));  // exact GELU
      }
      store_D_bf16(hmat, F_DIM, rt * 16, c * 16, hf);
    }
  }
  __syncthreads();

#pragma unroll
  for (int rr = 0; rr < 2; ++rr) {
    int rt = wave * 2 + rr;
    v8f a0 = bias_frag(b2, 0), a1 = bias_frag(b2, 16);
#pragma unroll
    for (int kc = 0; kc < 2; ++kc) {
      v16bf aH = load_A(hmat, F_DIM, rt * 16, kc * 32);
      a0 = wmma_bf16(aH, load_B(W2T, F_DIM, 0, kc * 32), a0);
      a1 = wmma_bf16(aH, load_B(W2T, F_DIM, 16, kc * 32), a1);
    }
    add_D_f32(tokf, E_DIM, rt * 16, 0, a0);
    add_D_f32(tokf, E_DIM, rt * 16, 16, a1);
  }
  __syncthreads();

  // ---- LN2 ----
  ln_row(tokf, nullptr, g2, be2, tid);
  __syncthreads();

  // ---- Stage E: mean-pool + output head ----
  {
    int e = tid & 31, grp = tid >> 5;
    float ps = 0.f;
#pragma unroll
    for (int i = 0; i < 32; ++i) ps += tokf[(grp * 32 + i) * E_DIM + e];
    poolscr[grp * 32 + e] = ps;
  }
  __syncthreads();
  if (tid < 32) {
    float tot = 0.f;
#pragma unroll
    for (int g = 0; g < 8; ++g) tot += poolscr[g * 32 + tid];
    pooled[tid] = tot * (1.f / S_LEN);
  }
  __syncthreads();
  if (tid < O_DIM) {
    float r = bo[tid];
#pragma unroll
    for (int e = 0; e < E_DIM; ++e) r += pooled[e] * wo[e * O_DIM + tid];
    out[b * O_DIM + tid] = r;
  }
}

extern "C" void kernel_launch(void* const* d_in, const int* in_sizes, int n_in,
                              void* d_out, int out_size, void* d_ws, size_t ws_size,
                              hipStream_t stream) {
  const float* x     = (const float*)d_in[0];
  const float* w_tok = (const float*)d_in[1];
  const float* b_tok = (const float*)d_in[2];
  const float* pos   = (const float*)d_in[3];
  const float* wq    = (const float*)d_in[4];
  const float* bq    = (const float*)d_in[5];
  const float* wk    = (const float*)d_in[6];
  const float* bk    = (const float*)d_in[7];
  const float* wv    = (const float*)d_in[8];
  const float* bv    = (const float*)d_in[9];
  const float* w1    = (const float*)d_in[10];
  const float* b1    = (const float*)d_in[11];
  const float* w2    = (const float*)d_in[12];
  const float* b2    = (const float*)d_in[13];
  const float* g1    = (const float*)d_in[14];
  const float* be1   = (const float*)d_in[15];
  const float* g2    = (const float*)d_in[16];
  const float* be2   = (const float*)d_in[17];
  const float* wo    = (const float*)d_in[18];
  const float* bo    = (const float*)d_in[19];

  bf16* ws = (bf16*)d_ws;
  int B = in_sizes[0] / S_LEN;  // 2048

  prep_weights<<<1, 256, 0, stream>>>(wq, wk, wv, w1, w2, ws);
  attn_block_kernel<<<B, 256, 0, stream>>>(x, w_tok, b_tok, pos, bq, bk, bv, b1, b2,
                                           g1, be1, g2, be2, wo, bo, ws,
                                           (float*)d_out);
}
